// TemporalWormhole_56762287784371
// MI455X (gfx1250) — matrix operand order repaired and checked by reading, not compiled, not run on hardware
//
#include <hip/hip_runtime.h>
#include <hip/hip_bf16.h>

// ---------------------------------------------------------------------------
// TemporalWormhole for MI455X (gfx1250): all dense GEMMs + windowed attention
// via v_wmma_f32_16x16x32_f16 (wave32 WMMA, f32 accumulate); global->LDS
// staging uses the CDNA5 async-to-LDS path (ASYNCcnt) when available, with
// double-buffered LDS tiles so the DMA overlaps WMMA issue.
// ---------------------------------------------------------------------------

typedef __attribute__((ext_vector_type(16))) _Float16 v16h;
typedef __attribute__((ext_vector_type(8)))  float    v8f;

#define TW_T 2048
#define TW_D 512
#define TW_B 2
#define TW_M 4096              // B*T rows per band

// ---- async global->LDS copy (CDNA5), with portable fallback ---------------

#if defined(__has_builtin)
#  if __has_builtin(__builtin_amdgcn_global_load_async_to_lds_b128)
#    define TW_HAS_ASYNC 1
#  endif
#endif
#ifndef TW_HAS_ASYNC
#  define TW_HAS_ASYNC 0
#endif

typedef int tw_v4i __attribute__((ext_vector_type(4)));
typedef __attribute__((address_space(1))) tw_v4i* tw_gp;   // global int4*
typedef __attribute__((address_space(3))) tw_v4i* tw_lp;   // LDS int4*

__device__ __forceinline__ void tw_cp16(_Float16* dst, const _Float16* src) {
#if TW_HAS_ASYNC
    __builtin_amdgcn_global_load_async_to_lds_b128(
        (tw_gp)(_Float16*)src, (tw_lp)dst, 0, 0);
#else
    *(uint4*)dst = *(const uint4*)src;
#endif
}

__device__ __forceinline__ void tw_wait_async() {
#if TW_HAS_ASYNC
#  if __has_builtin(__builtin_amdgcn_s_wait_asynccnt)
    __builtin_amdgcn_s_wait_asynccnt(0);
#  else
    asm volatile("s_wait_asynccnt 0" ::: "memory");
#  endif
#endif
}

// ---- WMMA helpers ---------------------------------------------------------

__device__ __forceinline__ v8f tw_wmma(v16h a, v16h b, v8f c) {
    return __builtin_amdgcn_wmma_f32_16x16x32_f16(
        /*neg_a=*/false, a, /*neg_b=*/false, b,
        /*c_mod=*/(short)0, c, /*reuse_a=*/false, /*reuse_b=*/false);
}

// A-matrix 16x32 f16 fragment (ISA 7.12.2): lane<16 holds K[hi*8..+7] and
// K[16+hi*8..+7]; pointer is pre-offset by hi*8 within the 32-wide K slice.
__device__ __forceinline__ v16h tw_frag_a(const _Float16* p) {
    union { v16h v; uint4 u[2]; } t;
    t.u[0] = *(const uint4*)(p);
    t.u[1] = *(const uint4*)(p + 16);
    return t.v;
}

// B-matrix 32x16 f16 fragment: lane holds column (lane&15), K[hi*8..+7] and
// K[16+hi*8..+7]; pointer pre-offset by hi*8.
__device__ __forceinline__ v16h tw_frag_b(const _Float16* p) {
    union { v16h v; uint4 u[2]; } t;
    t.u[0] = *(const uint4*)(p);
    t.u[1] = *(const uint4*)(p + 16);
    return t.v;
}

// ---- elementwise kernels --------------------------------------------------

__global__ void tw_cvt16(const float* __restrict__ in, _Float16* __restrict__ out, size_t n) {
    size_t i = (size_t)blockIdx.x * blockDim.x + threadIdx.x;
    if (i < n) out[i] = (_Float16)in[i];
}

// W (K x N) f32  ->  Wt (N x K) f16, batched over blockIdx.z
__global__ void tw_transpose16(const float* __restrict__ W, _Float16* __restrict__ Wt,
                               int K, int N) {
    __shared__ float tile[32][33];
    size_t ms = (size_t)K * N * blockIdx.z;
    const float* Wm = W + ms;
    _Float16* Wtm = Wt + ms;
    int n0 = blockIdx.x * 32, k0 = blockIdx.y * 32;
    int tx = threadIdx.x, ty = threadIdx.y;          // 32 x 8
    #pragma unroll
    for (int i = 0; i < 32; i += 8)
        tile[ty + i][tx] = Wm[(size_t)(k0 + ty + i) * N + n0 + tx];
    __syncthreads();
    #pragma unroll
    for (int i = 0; i < 32; i += 8)
        Wtm[(size_t)(n0 + ty + i) * K + k0 + tx] = (_Float16)tile[tx][ty + i];
}

__global__ void tw_meanctx(const _Float16* __restrict__ ctx, _Float16* __restrict__ octx,
                           size_t md) {
    size_t i = (size_t)blockIdx.x * blockDim.x + threadIdx.x;
    if (i >= md) return;
    float s = 0.f;
    #pragma unroll
    for (int n = 0; n < 7; ++n) s += (float)ctx[(size_t)n * md + i];
    s -= (float)ctx[(size_t)3 * md + i];
    octx[i] = (_Float16)(s * (1.0f / 6.0f));
}

__global__ void tw_final(const float* __restrict__ bands, const float* __restrict__ Z,
                         const float* __restrict__ worm, float* __restrict__ out, size_t n) {
    size_t i = (size_t)blockIdx.x * blockDim.x + threadIdx.x;
    if (i >= n) return;
    float g = 1.0f / (1.0f + __expf(-Z[i]));
    out[i] = bands[i] + g * worm[i];
}

// ---- tiled WMMA GEMM:  C(4096x512) = A_f16(M x K, lda) * Wt_f16(512 x K, ldw)^T
//      + bias(N) [+ Cin]; double-buffered async LDS staging. ------------------

__global__ __launch_bounds__(256)
void tw_gemm(const _Float16* __restrict__ A, int lda,
             const _Float16* __restrict__ Wt, int ldw,
             const float* __restrict__ bias,
             const float* __restrict__ Cin,
             float* __restrict__ C32,
             _Float16* __restrict__ C16,
             int K) {
    __shared__ __attribute__((aligned(16))) _Float16 sA[2][128][40];
    __shared__ __attribute__((aligned(16))) _Float16 sB[2][128][40];
    const int N = TW_D;
    int tid = threadIdx.x, lane = tid & 31, wave = tid >> 5;
    int wm = wave & 3, wn = wave >> 2;               // 4 x 2 wave grid
    int rowBase = blockIdx.x * 128, colBase = blockIdx.y * 128;
    int r = lane & 15, hi = lane >> 4;

    // fixed per-thread staging slots: chunks c = tid and tid+256 of 512
    int c0 = tid, c1 = tid + 256;
    int r0 = c0 >> 2, o0 = (c0 & 3) << 3;
    int r1 = c1 >> 2, o1 = (c1 & 3) << 3;
    const _Float16* gA0 = A  + (size_t)(rowBase + r0) * lda + o0;
    const _Float16* gA1 = A  + (size_t)(rowBase + r1) * lda + o1;
    const _Float16* gB0 = Wt + (size_t)(colBase + r0) * ldw + o0;
    const _Float16* gB1 = Wt + (size_t)(colBase + r1) * ldw + o1;

    v8f acc[2][4];
    #pragma unroll
    for (int i = 0; i < 2; ++i)
        #pragma unroll
        for (int j = 0; j < 4; ++j)
            acc[i][j] = (v8f){0.f,0.f,0.f,0.f,0.f,0.f,0.f,0.f};

    // prefetch first K-tile into buffer 0
    tw_cp16(&sA[0][r0][o0], gA0);
    tw_cp16(&sA[0][r1][o1], gA1);
    tw_cp16(&sB[0][r0][o0], gB0);
    tw_cp16(&sB[0][r1][o1], gB1);

    int buf = 0;
    for (int k0 = 0; k0 < K; k0 += 32) {
        tw_wait_async();
        __syncthreads();
        if (k0 + 32 < K) {                           // prefetch next tile
            int nb = buf ^ 1, kn = k0 + 32;
            tw_cp16(&sA[nb][r0][o0], gA0 + kn);
            tw_cp16(&sA[nb][r1][o1], gA1 + kn);
            tw_cp16(&sB[nb][r0][o0], gB0 + kn);
            tw_cp16(&sB[nb][r1][o1], gB1 + kn);
        }
        v16h af[2], bf[4];
        #pragma unroll
        for (int i = 0; i < 2; ++i) af[i] = tw_frag_a(&sA[buf][wm * 32 + i * 16 + r][hi * 8]);
        #pragma unroll
        for (int j = 0; j < 4; ++j) bf[j] = tw_frag_b(&sB[buf][wn * 64 + j * 16 + r][hi * 8]);
        #pragma unroll
        for (int i = 0; i < 2; ++i)
            #pragma unroll
            for (int j = 0; j < 4; ++j)
                acc[i][j] = tw_wmma(af[i], bf[j], acc[i][j]);
        buf ^= 1;
    }

    #pragma unroll
    for (int i = 0; i < 2; ++i) {
        int row0 = rowBase + wm * 32 + i * 16 + hi * 8;   // VGPR e -> M = e (+8 for hi lanes)
        #pragma unroll
        for (int j = 0; j < 4; ++j) {
            int col = colBase + wn * 64 + j * 16 + r;
            float bv = bias ? bias[col] : 0.f;
            #pragma unroll
            for (int e = 0; e < 8; ++e) {
                size_t idx = (size_t)(row0 + e) * N + col;
                float val = acc[i][j][e] + bv;
                if (Cin) val += Cin[idx];
                if (C32) C32[idx] = val;
                if (C16) C16[idx] = (_Float16)val;
            }
        }
    }
}

// ---- fused windowed attention: per-band, 16-query tile per workgroup ------
// scores = (Q K^T)/sqrt(D) over window, softmax, ctx = P V  (all WMMA f16).

__global__ __launch_bounds__(128)
void tw_attn(const _Float16* __restrict__ q,
             const _Float16* __restrict__ k,
             const _Float16* __restrict__ v,
             _Float16* __restrict__ ctx,
             int w, int Wp) {
    extern __shared__ __attribute__((aligned(16))) char smem[];
    _Float16* sQ   = (_Float16*)smem;                       // 16 x 512
    _Float16* slab = sQ + 16 * TW_D;                        // Wp x 512 (K) then 512 x Wp (V^T)
    float*    sSc  = (float*)(slab + (size_t)Wp * TW_D);    // 16 x Wp
    _Float16* sP   = (_Float16*)(sSc + 16 * Wp);            // 16 x Wp

    int tid = threadIdx.x, lane = tid & 31, wv = tid >> 5;  // 4 waves
    int b  = blockIdx.x >> 7;                               // T/16 = 128 tiles per batch
    int t0 = (blockIdx.x & 127) << 4;
    size_t bT = (size_t)b * TW_T;
    int base = t0 + 16 - Wp;                                // first key index of slab
    int r = lane & 15, hi = lane >> 4;

    // stage Q tile and K slab (clamped rows; masking handles invalids)
    for (int c = tid; c < 16 * 64; c += 128) {
        int rr = c >> 6, kk = (c & 63) << 3;
        tw_cp16(&sQ[rr * TW_D + kk], &q[(bT + t0 + rr) * TW_D + kk]);
    }
    int nch = Wp << 6;
    for (int c = tid; c < nch; c += 128) {
        int j = c >> 6, kk = (c & 63) << 3;
        int s = base + j; s = s < 0 ? 0 : s;
        tw_cp16(&slab[(size_t)j * TW_D + kk], &k[(bT + s) * TW_D + kk]);
    }
    tw_wait_async();
    __syncthreads();

    // scores: S(16 x Wp) = Q K^T * scale
    const float scale = 0.044194173824159216f;              // 1/sqrt(512)
    int ntile = Wp >> 4;
    for (int jt = wv; jt < ntile; jt += 4) {
        v8f acc = (v8f){0.f,0.f,0.f,0.f,0.f,0.f,0.f,0.f};
        for (int kb = 0; kb < TW_D; kb += 32) {
            v16h a = tw_frag_a(&sQ[r * TW_D + kb + hi * 8]);
            v16h bb = tw_frag_b(&slab[(size_t)(jt * 16 + r) * TW_D + kb + hi * 8]);
            acc = tw_wmma(a, bb, acc);
        }
        #pragma unroll
        for (int e = 0; e < 8; ++e)
            sSc[(e + hi * 8) * Wp + jt * 16 + r] = acc[e] * scale;
    }
    __syncthreads();

    // windowed softmax (equivalent to -10000 mask + full softmax)
    if (tid < 16) {
        int t = t0 + tid;
        float m = -1e30f;
        for (int j = 0; j < Wp; ++j) {
            int s = base + j;
            if (s >= 0 && s <= t && (t - s) < w) {
                float x = sSc[tid * Wp + j];
                m = x > m ? x : m;
            }
        }
        float sum = 0.f;
        for (int j = 0; j < Wp; ++j) {
            int s = base + j;
            bool ok = (s >= 0) && (s <= t) && ((t - s) < w);
            float e = ok ? __expf(sSc[tid * Wp + j] - m) : 0.f;
            sSc[tid * Wp + j] = e;
            sum += e;
        }
        float inv = 1.f / sum;
        for (int j = 0; j < Wp; ++j)
            sP[tid * Wp + j] = (_Float16)(sSc[tid * Wp + j] * inv);
    }
    __syncthreads();

    // restage V transposed (through VGPRs): slab becomes V^T (512 x Wp)
    for (int c = tid; c < nch; c += 128) {
        int j = c >> 6, d0 = (c & 63) << 3;
        int s = base + j; s = s < 0 ? 0 : s;
        uint4 raw = *(const uint4*)&v[(bT + s) * TW_D + d0];
        const _Float16* h = (const _Float16*)&raw;
        #pragma unroll
        for (int i = 0; i < 8; ++i) slab[(size_t)(d0 + i) * Wp + j] = h[i];
    }
    __syncthreads();

    // ctx(16 x 512) = P(16 x Wp) @ V(Wp x 512)
    for (int nt = wv; nt < 32; nt += 4) {
        v8f acc = (v8f){0.f,0.f,0.f,0.f,0.f,0.f,0.f,0.f};
        for (int kb = 0; kb < Wp; kb += 32) {
            v16h a = tw_frag_a(&sP[r * Wp + kb + hi * 8]);
            v16h bb = tw_frag_b(&slab[(size_t)(nt * 16 + r) * Wp + kb + hi * 8]);
            acc = tw_wmma(a, bb, acc);
        }
        #pragma unroll
        for (int e = 0; e < 8; ++e) {
            int row = t0 + e + hi * 8;
            ctx[(bT + row) * TW_D + nt * 16 + r] = (_Float16)acc[e];
        }
    }
}

// ---------------------------------------------------------------------------

extern "C" void kernel_launch(void* const* d_in, const int* in_sizes, int n_in,
                              void* d_out, int out_size, void* d_ws, size_t ws_size,
                              hipStream_t stream) {
    const float* bands   = (const float*)d_in[0];
    const float* qW      = (const float*)d_in[1];
    const float* qb      = (const float*)d_in[2];
    const float* kW      = (const float*)d_in[3];
    const float* kb      = (const float*)d_in[4];
    const float* vW      = (const float*)d_in[5];
    const float* vb      = (const float*)d_in[6];
    const float* crossW  = (const float*)d_in[7];
    const float* crossb  = (const float*)d_in[8];
    const float* crossrW = (const float*)d_in[9];
    const float* crossrb = (const float*)d_in[10];
    const float* gateW   = (const float*)d_in[11];
    const float* gateb   = (const float*)d_in[12];
    const float* bridgeW = (const float*)d_in[13];
    const float* bridgeb = (const float*)d_in[14];
    float* out = (float*)d_out;

    const size_t MD = (size_t)TW_M * TW_D;       // 2,097,152 per band
    const size_t DD = (size_t)TW_D * TW_D;

    char* ws = (char*)d_ws;
    size_t off = 0;
    auto alloc = [&](size_t bytes) -> char* {
        char* p = ws + off;
        off += (bytes + 255) & ~(size_t)255;
        return p;
    };
    _Float16* bands16  = (_Float16*)alloc(7 * MD * 2);
    _Float16* q16      = (_Float16*)alloc(7 * MD * 2);
    _Float16* k16      = (_Float16*)alloc(7 * MD * 2);
    _Float16* v16f     = (_Float16*)alloc(7 * MD * 2);
    _Float16* ctx16    = (_Float16*)alloc(7 * MD * 2);
    _Float16* octx16   = (_Float16*)alloc(MD * 2);
    _Float16* worm16   = (_Float16*)alloc(7 * MD * 2);
    float*    worm32   = (float*)   alloc(7 * MD * 4);
    float*    Z32      = (float*)   alloc(7 * MD * 4);
    _Float16* qWt      = (_Float16*)alloc(7 * DD * 2);
    _Float16* kWt      = (_Float16*)alloc(7 * DD * 2);
    _Float16* vWt      = (_Float16*)alloc(7 * DD * 2);
    _Float16* crossWt  = (_Float16*)alloc(3 * DD * 2);
    _Float16* crossrWt = (_Float16*)alloc(3 * DD * 2);
    _Float16* gateWt   = (_Float16*)alloc(7 * 2 * DD * 2);
    _Float16* bridgeWt = (_Float16*)alloc(DD * 2);

    // 1) convert activations, transpose+convert weights
    {
        size_t n = 7 * MD;
        tw_cvt16<<<dim3((unsigned)(n / 256)), dim3(256), 0, stream>>>(bands, bands16, n);
    }
    tw_transpose16<<<dim3(16, 16, 7), dim3(32, 8), 0, stream>>>(qW, qWt, 512, 512);
    tw_transpose16<<<dim3(16, 16, 7), dim3(32, 8), 0, stream>>>(kW, kWt, 512, 512);
    tw_transpose16<<<dim3(16, 16, 7), dim3(32, 8), 0, stream>>>(vW, vWt, 512, 512);
    tw_transpose16<<<dim3(16, 16, 3), dim3(32, 8), 0, stream>>>(crossW, crossWt, 512, 512);
    tw_transpose16<<<dim3(16, 16, 3), dim3(32, 8), 0, stream>>>(crossrW, crossrWt, 512, 512);
    tw_transpose16<<<dim3(16, 32, 7), dim3(32, 8), 0, stream>>>(gateW, gateWt, 1024, 512);
    tw_transpose16<<<dim3(16, 16, 1), dim3(32, 8), 0, stream>>>(bridgeW, bridgeWt, 512, 512);

    dim3 ggrid(TW_M / 128, TW_D / 128);   // 32 x 4
    auto gemm = [&](const _Float16* A, int lda, const _Float16* Wt, int ldw,
                    const float* bias, const float* Cin, float* C32, _Float16* C16) {
        tw_gemm<<<ggrid, dim3(256), 0, stream>>>(A, lda, Wt, ldw, bias, Cin, C32, C16, 512);
    };

    // 2) q,k,v projections (f16 outputs for attention)
    for (int n = 0; n < 7; ++n) {
        gemm(bands16 + n * MD, 512, qWt + n * DD, 512, qb + n * 512, nullptr, nullptr, q16 + n * MD);
        gemm(bands16 + n * MD, 512, kWt + n * DD, 512, kb + n * 512, nullptr, nullptr, k16 + n * MD);
        gemm(bands16 + n * MD, 512, vWt + n * DD, 512, vb + n * 512, nullptr, nullptr, v16f + n * MD);
    }

    // 3) windowed attention per band
    const int win[7] = {128, 64, 32, 16, 16, 8, 4};
    const int wp[7]  = {160, 96, 64, 32, 32, 32, 32};   // ceil((w+15)/32)*32
    for (int n = 0; n < 7; ++n) {
        size_t shmem = (size_t)16 * TW_D * 2 + (size_t)wp[n] * TW_D * 2
                     + (size_t)16 * wp[n] * 4 + (size_t)16 * wp[n] * 2;
        tw_attn<<<dim3(TW_B * TW_T / 16), dim3(128), shmem, stream>>>(
            q16 + n * MD, k16 + n * MD, v16f + n * MD, ctx16 + n * MD, win[n], wp[n]);
    }

    // 4) cross-band wormholes, mapped to destination band:
    //    worm[6,5,4] = ctx[0,1,2] @ crossW[p];  worm[0,1,2] = ctx[6,5,4] @ crossrW[p]
    const int plow[3] = {0, 1, 2}, phigh[3] = {6, 5, 4};
    for (int p = 0; p < 3; ++p) {
        int dh = phigh[p], dl = plow[p];
        gemm(ctx16 + plow[p] * MD, 512, crossWt + p * DD, 512, crossb + p * 512,
             nullptr, worm32 + dh * MD, worm16 + dh * MD);
        gemm(ctx16 + phigh[p] * MD, 512, crossrWt + p * DD, 512, crossrb + p * 512,
             nullptr, worm32 + dl * MD, worm16 + dl * MD);
    }

    // 5) bridge band 3: worm[3] = mean(other ctx) @ bridgeW
    tw_meanctx<<<dim3((unsigned)(MD / 256)), dim3(256), 0, stream>>>(ctx16, octx16, MD);
    gemm(octx16, 512, bridgeWt, 512, bridgeb, nullptr, worm32 + 3 * MD, worm16 + 3 * MD);

    // 6) gate logits: Z[n] = bands[n]@G[n][:D] + worm[n]@G[n][D:] + gb[n]
    for (int n = 0; n < 7; ++n) {
        gemm(bands16 + n * MD, 512, gateWt + (size_t)n * 2 * DD, 1024, gateb + n * 512,
             nullptr, Z32 + n * MD, nullptr);
        gemm(worm16 + n * MD, 512, gateWt + (size_t)n * 2 * DD + 512, 1024, nullptr,
             Z32 + n * MD, Z32 + n * MD, nullptr);
    }

    // 7) out = bands + sigmoid(Z) * worm
    {
        size_t n = 7 * MD;
        tw_final<<<dim3((unsigned)(n / 256)), dim3(256), 0, stream>>>(bands, Z32, worm32, out, n);
    }
}
